// BitLinear_49031346651240
// MI455X (gfx1250) — matrix operand order repaired and checked by reading, not compiled, not run on hardware
//
#include <hip/hip_runtime.h>

// ---------------------------------------------------------------------------
// BitLinear: out = input @ (sign(W)*mean|W|)^T + bias
// TOKENS=8192, D_IN=2048, D_OUT=8192, fp32 in/out.
// GEMM path: bf16 x (+-1 bf16) -> f32 via v_wmma_f32_16x16x32_bf16 (wave32),
// double-buffered LDS tiles filled by GLOBAL_LOAD_ASYNC_TO_LDS_B128 (ASYNCcnt)
// so the tensor DMA overlaps the matrix pipe.
// ---------------------------------------------------------------------------

#define TOKENS 8192
#define D_IN   2048
#define D_OUT  8192

// Flip to 0 if the async-to-LDS inline asm is rejected by the assembler.
#define USE_ASYNC_LDS 1

typedef __attribute__((ext_vector_type(16))) __bf16 v16bf;
typedef __attribute__((ext_vector_type(8)))  float  v8f;

// round-to-nearest-even fp32 -> bf16 (bit pattern), no __bf16 arithmetic
__device__ __forceinline__ unsigned short f32_to_bf16_rne(float f) {
    unsigned int u = __float_as_uint(f);
    u += 0x7FFFu + ((u >> 16) & 1u);
    return (unsigned short)(u >> 16);
}

// ---------------- scale = mean(|W|), two-pass deterministic ----------------
__global__ void absmean_partial(const float* __restrict__ W,
                                float* __restrict__ part, int n) {
    __shared__ float sm[256];
    float s = 0.f;
    for (int i = blockIdx.x * blockDim.x + threadIdx.x; i < n;
         i += gridDim.x * blockDim.x)
        s += fabsf(W[i]);
    sm[threadIdx.x] = s;
    __syncthreads();
    for (int off = 128; off > 0; off >>= 1) {
        if ((int)threadIdx.x < off) sm[threadIdx.x] += sm[threadIdx.x + off];
        __syncthreads();
    }
    if (threadIdx.x == 0) part[blockIdx.x] = sm[0];
}

__global__ void absmean_final(const float* __restrict__ part,
                              float* __restrict__ scale, int nparts, float inv_n) {
    __shared__ float sm[256];
    float s = 0.f;
    for (int i = threadIdx.x; i < nparts; i += 256) s += part[i];
    sm[threadIdx.x] = s;
    __syncthreads();
    for (int off = 128; off > 0; off >>= 1) {
        if ((int)threadIdx.x < off) sm[threadIdx.x] += sm[threadIdx.x + off];
        __syncthreads();
    }
    if (threadIdx.x == 0) *scale = sm[0] * inv_n;
}

// ---------------- fp32 -> bf16 conversion of the activations ----------------
__global__ void cvt_a(const float* __restrict__ A,
                      unsigned short* __restrict__ Ab, int n4) {
    for (int i = blockIdx.x * blockDim.x + threadIdx.x; i < n4;
         i += gridDim.x * blockDim.x) {
        float4 v = ((const float4*)A)[i];
        ushort4 o;
        o.x = f32_to_bf16_rne(v.x);
        o.y = f32_to_bf16_rne(v.y);
        o.z = f32_to_bf16_rne(v.z);
        o.w = f32_to_bf16_rne(v.w);
        ((ushort4*)Ab)[i] = o;
    }
}

// ---------------- fp32 W -> sign(W) as exact bf16 {-1,0,+1} ----------------
__global__ void quant_w(const float* __restrict__ W,
                        unsigned short* __restrict__ Wq, int n4) {
    for (int i = blockIdx.x * blockDim.x + threadIdx.x; i < n4;
         i += gridDim.x * blockDim.x) {
        float4 v = ((const float4*)W)[i];
        ushort4 o;
        o.x = (v.x > 0.f) ? 0x3F80u : ((v.x < 0.f) ? 0xBF80u : 0u);
        o.y = (v.y > 0.f) ? 0x3F80u : ((v.y < 0.f) ? 0xBF80u : 0u);
        o.z = (v.z > 0.f) ? 0x3F80u : ((v.z < 0.f) ? 0xBF80u : 0u);
        o.w = (v.w > 0.f) ? 0x3F80u : ((v.w < 0.f) ? 0xBF80u : 0u);
        ((ushort4*)Wq)[i] = o;
    }
}

// ---------------- WMMA GEMM ----------------
// Block tile 128(M) x 128(N), BK = 64 (two 16x16x32 WMMA K-substeps/stage).
// 256 threads = 8 waves (wave32). Wave grid 2(M) x 4(N): each wave owns a
// 64x32 tile = 4x2 accumulators of 16x16 f32. LDS double buffered.
#define BM 128
#define BN 128
#define BK 64
#define BKP 72   // padded row: 144B stride = 9*16B (aligned, bank-conflict-free)

// one 16B chunk global -> LDS
__device__ __forceinline__ void cp16(const unsigned short* g, unsigned short* l) {
#if USE_ASYNC_LDS
    unsigned lds = (unsigned)(uintptr_t)l;   // aperture low bits = LDS byte addr
    asm volatile("global_load_async_to_lds_b128 %0, %1, off"
                 :: "v"(lds), "v"(g) : "memory");
#else
    *(uint4*)l = *(const uint4*)g;
#endif
}

__device__ __forceinline__ void wait_async() {
#if USE_ASYNC_LDS
#if __has_builtin(__builtin_amdgcn_s_wait_asynccnt)
    __builtin_amdgcn_s_wait_asynccnt(0);
#else
    asm volatile("s_wait_asynccnt 0x0" ::: "memory");
#endif
#endif
}

// A-fragment, 16-bit A 16x32 per ISA table:
// lanes 0-15 hold K = 0..7 then 16..23 ; lanes 16-31 hold K = 8..15 then 24..31
__device__ __forceinline__ v16bf load_fragA(const unsigned short* rowbase, int kbase) {
    union { uint4 u[2]; v16bf v; } r;
    r.u[0] = *(const uint4*)(rowbase + kbase);
    r.u[1] = *(const uint4*)(rowbase + kbase + 16);
    return r.v;
}

// B-fragment: lane holds column n = lane%16; lanes 0-15 hold K = 0..7 then 16..23,
// lanes 16-31 hold K = 8..15 then 24..31 (same as A).
__device__ __forceinline__ v16bf load_fragB(const unsigned short* rowbase, int koff) {
    union { uint4 u[2]; v16bf v; } r;
    r.u[0] = *(const uint4*)(rowbase + koff);
    r.u[1] = *(const uint4*)(rowbase + koff + 16);
    return r.v;
}

__global__ void __launch_bounds__(256, 1)
bitlinear_gemm(const unsigned short* __restrict__ Ab,  // [TOKENS][D_IN] bf16
               const unsigned short* __restrict__ Wq,  // [D_OUT][D_IN] bf16 +-1
               const float* __restrict__ bias,         // [D_OUT]
               const float* __restrict__ scalep,       // scalar
               float* __restrict__ out)                // [TOKENS][D_OUT]
{
    __shared__ unsigned short As[2][BM][BKP];
    __shared__ unsigned short Bs[2][BN][BKP];

    const int tid  = threadIdx.x;
    const int lane = tid & 31;
    const int wave = tid >> 5;        // 0..7
    const int wm   = wave & 1;        // M slot: 0/1 -> 64 rows each
    const int wn   = wave >> 1;       // N slot: 0..3 -> 32 cols each

    const int blockM = blockIdx.y * BM;
    const int blockN = blockIdx.x * BN;

    // staging: tile is 128 rows x 64 bf16 = 1024 x 16B chunks; 256 threads ->
    // 4 chunks each per matrix. chunk c covers row (tid/8 + 32*c), seg tid%8.
    const int srow = tid >> 3;          // 0..31
    const int sseg = (tid & 7) * 8;     // bf16 elem offset within row (16B units)

    const unsigned short* gA = Ab + (size_t)(blockM + srow) * D_IN + sseg;
    const unsigned short* gB = Wq + (size_t)(blockN + srow) * D_IN + sseg;

    const float scale = *scalep;

    v8f acc[4][2];
#pragma unroll
    for (int mi = 0; mi < 4; ++mi)
#pragma unroll
        for (int ni = 0; ni < 2; ++ni)
#pragma unroll
            for (int e = 0; e < 8; ++e) acc[mi][ni][e] = 0.f;

    const int laneM  = lane & 15;
    const int kbaseA = (lane < 16) ? 0 : 8;   // A-frag K split
    const int koffB  = (lane < 16) ? 0 : 8;   // B-frag K split

    const int NKT = D_IN / BK;   // 32 K-tiles

    // ---- prologue: stage tile 0 into buffer 0 ----
#pragma unroll
    for (int c = 0; c < 4; ++c) {
        cp16(gA + (size_t)(32 * c) * D_IN, &As[0][srow + 32 * c][sseg]);
        cp16(gB + (size_t)(32 * c) * D_IN, &Bs[0][srow + 32 * c][sseg]);
    }
    wait_async();
    __syncthreads();

    for (int kt = 0; kt < NKT; ++kt) {
        const int cur = kt & 1;
        const int nxt = cur ^ 1;

        // ---- issue async copy of next tile (overlaps the WMMAs below) ----
        if (kt + 1 < NKT) {
            const int k0 = (kt + 1) * BK;
#pragma unroll
            for (int c = 0; c < 4; ++c) {
                cp16(gA + (size_t)(32 * c) * D_IN + k0,
                     &As[nxt][srow + 32 * c][sseg]);
                cp16(gB + (size_t)(32 * c) * D_IN + k0,
                     &Bs[nxt][srow + 32 * c][sseg]);
            }
        }

        // ---- compute current tile: 2 K-substeps x 8 WMMAs ----
#pragma unroll
        for (int ks = 0; ks < 2; ++ks) {
            const int kb = ks * 32;
            v16bf afrag[4], bfrag[2];
#pragma unroll
            for (int mi = 0; mi < 4; ++mi) {
                const int m = wm * 64 + mi * 16 + laneM;
                afrag[mi] = load_fragA(&As[cur][m][kb], kbaseA);
            }
#pragma unroll
            for (int ni = 0; ni < 2; ++ni) {
                const int n = wn * 32 + ni * 16 + laneM;
                bfrag[ni] = load_fragB(&Bs[cur][n][kb], koffB);
            }
#pragma unroll
            for (int mi = 0; mi < 4; ++mi)
#pragma unroll
                for (int ni = 0; ni < 2; ++ni)
                    acc[mi][ni] = __builtin_amdgcn_wmma_f32_16x16x32_bf16(
                        false, afrag[mi], false, bfrag[ni],
                        (short)0, acc[mi][ni], false, false);
        }

        // next buffer fully landed + everyone done reading cur before overwrite
        wait_async();
        __syncthreads();
    }

    // epilogue: C 16x16 f32 layout -> VGPR v holds M = v + (lane>=16 ? 8 : 0),
    // N = lane % 16 ; out = scale*acc + bias
    const int crow = (lane >> 4) << 3;   // 0 or 8
    const int ccol = lane & 15;
#pragma unroll
    for (int mi = 0; mi < 4; ++mi) {
#pragma unroll
        for (int ni = 0; ni < 2; ++ni) {
            const int o  = blockN + wn * 32 + ni * 16 + ccol;
            const float bv = bias[o];
            const int mBase = blockM + wm * 64 + mi * 16 + crow;
#pragma unroll
            for (int v = 0; v < 8; ++v)
                out[(size_t)(mBase + v) * D_OUT + o] = scale * acc[mi][ni][v] + bv;
        }
    }
}

// ---------------------------------------------------------------------------
extern "C" void kernel_launch(void* const* d_in, const int* in_sizes, int n_in,
                              void* d_out, int out_size, void* d_ws, size_t ws_size,
                              hipStream_t stream) {
    const float* inp    = (const float*)d_in[0];   // [TOKENS, D_IN]
    const float* weight = (const float*)d_in[1];   // [D_OUT, D_IN]
    const float* bias   = (const float*)d_in[2];   // [D_OUT]
    float* out = (float*)d_out;

    // workspace layout: bf16 A (32MB) | bf16 Wq (32MB) | scale + partials
    unsigned short* Ab = (unsigned short*)d_ws;
    unsigned short* Wq = (unsigned short*)((char*)d_ws + (size_t)TOKENS * D_IN * 2);
    float* red = (float*)((char*)d_ws + (size_t)TOKENS * D_IN * 2
                                      + (size_t)D_OUT * D_IN * 2);
    float* scalep = red;       // red[0]
    float* parts  = red + 1;   // 1024 partials

    const int nW = D_OUT * D_IN;            // 16,777,216
    absmean_partial<<<1024, 256, 0, stream>>>(weight, parts, nW);
    absmean_final<<<1, 256, 0, stream>>>(parts, scalep, 1024, 1.0f / (float)nW);

    cvt_a<<<2048, 256, 0, stream>>>(inp, Ab, (TOKENS * D_IN) / 4);
    quant_w<<<2048, 256, 0, stream>>>(weight, Wq, nW / 4);

    dim3 grid(D_OUT / BN, TOKENS / BM);     // 64 x 64 blocks
    bitlinear_gemm<<<grid, 256, 0, stream>>>(Ab, Wq, bias, scalep, out);
}
